// MultiHeadAttention_73675868996499
// MI455X (gfx1250) — matrix operand order repaired and checked by reading, not compiled, not run on hardware
//
#include <hip/hip_runtime.h>

typedef _Float16 half_t;
typedef __attribute__((ext_vector_type(16))) _Float16 v16h;
typedef __attribute__((ext_vector_type(8)))  _Float16 v8h;
typedef __attribute__((ext_vector_type(8)))  float    v8f;

#define NHEAD  4
#define DK     32
#define DMODEL 512
#define HDIM   128      // NHEAD*DK
#define BATCH  8
#define SEQ    2048
#define ROWS   (BATCH*SEQ)   // 16384
#define INV_SQRT_DK 0.17677669529663687f

#if __has_builtin(__builtin_amdgcn_global_load_async_to_lds_b128)
#define ASYNC_LDS 1
#else
#define ASYNC_LDS 0
#endif

// async builtin operand types: non-const int4 pointers, global / LDS addrspace
typedef int v4i __attribute__((vector_size(4 * sizeof(int))));
typedef __attribute__((address_space(1))) v4i gbl_v4i;
typedef __attribute__((address_space(3))) v4i lds_v4i;

// ---------------------------------------------------------------------------
// helpers
// ---------------------------------------------------------------------------
__device__ __forceinline__ v16h pack16(v8h lo, v8h hi) {
  v16h r;
#pragma unroll
  for (int i = 0; i < 8; ++i) { r[i] = lo[i]; r[i + 8] = hi[i]; }
  return r;
}

__device__ __forceinline__ v8f wmma_f16(v16h a, v16h b, v8f c) {
  return __builtin_amdgcn_wmma_f32_16x16x32_f16(false, a, false, b, (short)0, c,
                                                false, false);
}

__device__ __forceinline__ void wait_async_le2() {
#if ASYNC_LDS
#if __has_builtin(__builtin_amdgcn_s_wait_asynccnt)
  __builtin_amdgcn_s_wait_asynccnt(2);
#else
  asm volatile("s_wait_asynccnt 0x2" ::: "memory");
#endif
#endif
}
__device__ __forceinline__ void wait_async_le0() {
#if ASYNC_LDS
#if __has_builtin(__builtin_amdgcn_s_wait_asynccnt)
  __builtin_amdgcn_s_wait_asynccnt(0);
#else
  asm volatile("s_wait_asynccnt 0x0" ::: "memory");
#endif
#endif
}

// ---------------------------------------------------------------------------
// 0a) f32 -> f16 elementwise convert (activations)
// ---------------------------------------------------------------------------
__global__ void f32_to_f16_kernel(const float* __restrict__ src,
                                  half_t* __restrict__ dst, int n) {
  int i = (blockIdx.x * blockDim.x + threadIdx.x) * 8;
  if (i + 8 <= n) {
    float4 a = *(const float4*)(src + i);
    float4 b = *(const float4*)(src + i + 4);
    half_t o[8] = {(half_t)a.x, (half_t)a.y, (half_t)a.z, (half_t)a.w,
                   (half_t)b.x, (half_t)b.y, (half_t)b.z, (half_t)b.w};
#pragma unroll
    for (int j = 0; j < 8; ++j) dst[i + j] = o[j];
  }
}

// ---------------------------------------------------------------------------
// 0b) weight transpose + convert: src [K,N] f32 row-major -> dst [N,K] f16
// ---------------------------------------------------------------------------
__global__ void w_transpose_kernel(const float* __restrict__ src,
                                   half_t* __restrict__ dst, int K, int N) {
  int idx = blockIdx.x * blockDim.x + threadIdx.x;   // idx = n*K + k
  if (idx < K * N) {
    int n = idx / K;
    int k = idx - n * K;
    dst[idx] = (half_t)src[k * N + n];
  }
}

// ---------------------------------------------------------------------------
// 1) fused QKV projection.  One wave -> one 16x64 output strip (A reused x4).
//    grid = (ROWS/16, HDIM/64, 3), block = 32.
//    z==0 -> Q (scaled by 1/sqrt(dk)) stored [b,h,S,dk]
//    z==1 -> K stored [b,h,S,dk]
//    z==2 -> V stored transposed [b,h,dk,S]
// ---------------------------------------------------------------------------
__global__ __launch_bounds__(32) void qkv_proj_kernel(
    const half_t* __restrict__ xq, const half_t* __restrict__ xk,
    const half_t* __restrict__ xv, const half_t* __restrict__ wtq,
    const half_t* __restrict__ wtk, const half_t* __restrict__ wtv,
    half_t* __restrict__ q16, half_t* __restrict__ k16,
    half_t* __restrict__ vT16) {
  const int z = blockIdx.z;
  const half_t* X  = (z == 0) ? xq  : (z == 1) ? xk  : xv;
  const half_t* WT = (z == 0) ? wtq : (z == 1) ? wtk : wtv;

  const int m0 = blockIdx.x * 16;
  const int nbase = blockIdx.y * 64;
  const int lane = threadIdx.x;
  const int lo = lane & 15, hi = lane >> 4;

  v8f acc[4];
#pragma unroll
  for (int t = 0; t < 4; ++t)
#pragma unroll
    for (int r = 0; r < 8; ++r) acc[t][r] = 0.f;

  const half_t* arow = X + (size_t)(m0 + lo) * DMODEL + hi * 16;

#pragma unroll 2
  for (int kk = 0; kk < DMODEL; kk += 32) {
    v16h a = pack16(*(const v8h*)(arow + kk), *(const v8h*)(arow + kk + 8));
#pragma unroll
    for (int t = 0; t < 4; ++t) {
      const half_t* bp =
          WT + (size_t)(nbase + t * 16 + lo) * DMODEL + kk + hi * 16;
      acc[t] = wmma_f16(a, pack16(*(const v8h*)bp, *(const v8h*)(bp + 8)),
                        acc[t]);
    }
  }

  const float scale = (z == 0) ? INV_SQRT_DK : 1.0f;
#pragma unroll
  for (int t = 0; t < 4; ++t) {
    const int n = nbase + t * 16 + lo;
    const int h = n >> 5;        // head
    const int d = n & 31;        // dk index
#pragma unroll
    for (int r = 0; r < 8; ++r) {
      int m = m0 + r + hi * 8;
      int b = m >> 11;           // SEQ = 2048
      int s = m & 2047;
      int bh = b * NHEAD + h;
      half_t v = (half_t)(acc[t][r] * scale);
      if (z == 2)
        vT16[((size_t)bh * DK + d) * SEQ + s] = v;
      else if (z == 0)
        q16[((size_t)bh * SEQ + s) * DK + d] = v;
      else
        k16[((size_t)bh * SEQ + s) * DK + d] = v;
    }
  }
}

// ---------------------------------------------------------------------------
// 2) flash attention.  Block = 4 waves = 64 query rows; the 32-key K/V tile is
//    staged into LDS once per block (async global->LDS when available, double
//    buffered) and all 4 waves read their WMMA fragments from LDS.
//    grid = (SEQ/64, BATCH*NHEAD), block = 128.
// ---------------------------------------------------------------------------
__global__ __launch_bounds__(128) void flash_attn_kernel(
    const half_t* __restrict__ q16, const half_t* __restrict__ k16,
    const half_t* __restrict__ vT16, half_t* __restrict__ ctx16) {
  __shared__ __align__(16) half_t ksh[2][32 * DK];   // 2 x 2KB key tiles
  __shared__ __align__(16) half_t vsh[2][32 * DK];   // 2 x 2KB value tiles
  __shared__ __align__(16) half_t pshm[4][16 * DK];  // per-wave P transpose

  const int bh = blockIdx.y;
  const int b = bh >> 2;       // NHEAD = 4
  const int h = bh & 3;
  const int tid = threadIdx.x;
  const int wave = tid >> 5;
  const int lane = tid & 31;
  const int lo = lane & 15, hi = lane >> 4;
  const int q0 = blockIdx.x * 64 + wave * 16;

  const half_t* ksrc = k16 + (size_t)bh * SEQ * DK;   // [S][dk] contiguous
  const half_t* vsrc = vT16 + (size_t)bh * DK * SEQ;  // [dk][S]

  // stage one 32-key K tile (contiguous 2KB) + V tile (32 rows x 64B)
  auto stage = [&](int kt, int buf) {
    const half_t* kg = ksrc + (size_t)kt * DK + tid * 8;     // 128 x 16B
    half_t* kl = &ksh[buf][tid * 8];
    const int vr = tid >> 2, vc = (tid & 3) * 8;             // 128 x 16B
    const half_t* vg = vsrc + (size_t)vr * SEQ + kt + vc;
    half_t* vl = &vsh[buf][vr * DK + vc];
#if ASYNC_LDS
    __builtin_amdgcn_global_load_async_to_lds_b128(
        (gbl_v4i*)kg, (lds_v4i*)kl, 0, 0);
    __builtin_amdgcn_global_load_async_to_lds_b128(
        (gbl_v4i*)vg, (lds_v4i*)vl, 0, 0);
#else
    *(v8h*)kl = *(const v8h*)kg;
    *(v8h*)vl = *(const v8h*)vg;
#endif
  };

  // Q A-fragment (pre-scaled by 1/sqrt(dk)); dk==32 == one WMMA K-step
  const half_t* qp = q16 + ((size_t)bh * SEQ + q0 + lo) * DK + hi * 16;
  const v16h aq = pack16(*(const v8h*)qp, *(const v8h*)(qp + 8));

  v8f ctx0, ctx1;
  float mrow[8], lrow[8];
#pragma unroll
  for (int r = 0; r < 8; ++r) {
    ctx0[r] = 0.f; ctx1[r] = 0.f;
    mrow[r] = -1e30f; lrow[r] = 0.f;
  }

  stage(0, 0);

  for (int kt = 0; kt < SEQ; kt += 32) {
    const int buf = (kt >> 5) & 1;
    if (kt + 32 < SEQ) {
      stage(kt + 32, buf ^ 1);   // prefetch next tile into other buffer
      wait_async_le2();          // current tile's copies complete
    } else {
      wait_async_le0();
    }
    __syncthreads();             // tile visible to all waves

    // ---- scores: S[16q x 32k] = Q x K^T, two N=16 WMMAs (K from LDS) ----
    const half_t* kp0 = &ksh[buf][lo * DK + hi * 16];
    const half_t* kp1 = &ksh[buf][(16 + lo) * DK + hi * 16];
    v16h bk0 = pack16(*(const v8h*)kp0, *(const v8h*)(kp0 + 8));
    v16h bk1 = pack16(*(const v8h*)kp1, *(const v8h*)(kp1 + 8));
    v8f z8;
#pragma unroll
    for (int r = 0; r < 8; ++r) z8[r] = 0.f;
    v8f s0 = wmma_f16(aq, bk0, z8);
    v8f s1 = wmma_f16(aq, bk1, z8);

    // ---- online softmax (row stats live in the same lanes as ctx rows) --
#pragma unroll
    for (int r = 0; r < 8; ++r) {
      float tm = fmaxf(s0[r], s1[r]);
#pragma unroll
      for (int off = 1; off < 16; off <<= 1)
        tm = fmaxf(tm, __shfl_xor(tm, off, 16));
      float mn = fmaxf(mrow[r], tm);
      float alpha = __expf(mrow[r] - mn);
      float p0 = __expf(s0[r] - mn);
      float p1 = __expf(s1[r] - mn);
      s0[r] = p0;
      s1[r] = p1;
      float rs = p0 + p1;
#pragma unroll
      for (int off = 1; off < 16; off <<= 1)
        rs += __shfl_xor(rs, off, 16);
      lrow[r] = lrow[r] * alpha + rs;
      mrow[r] = mn;
      ctx0[r] *= alpha;
      ctx1[r] *= alpha;
    }

    // ---- P: C-layout -> A-layout transpose through per-wave LDS ---------
#pragma unroll
    for (int r = 0; r < 8; ++r) {
      int row = r + hi * 8;
      pshm[wave][row * DK + lo]      = (half_t)s0[r];
      pshm[wave][row * DK + 16 + lo] = (half_t)s1[r];
    }
    __syncthreads();
    const half_t* pp = &pshm[wave][lo * DK + hi * 16];
    v16h ap = pack16(*(const v8h*)pp, *(const v8h*)(pp + 8));

    // ---- ctx += P x V, two N=16 WMMAs (V from LDS) ----------------------
    const half_t* vp0 = &vsh[buf][lo * DK + hi * 16];
    const half_t* vp1 = &vsh[buf][(16 + lo) * DK + hi * 16];
    v16h bv0 = pack16(*(const v8h*)vp0, *(const v8h*)(vp0 + 8));
    v16h bv1 = pack16(*(const v8h*)vp1, *(const v8h*)(vp1 + 8));
    ctx0 = wmma_f16(ap, bv0, ctx0);
    ctx1 = wmma_f16(ap, bv1, ctx1);

    __syncthreads();             // WAR: reads done before buf is re-staged
  }

  // ---- finalize: divide by l, store to ctx16 [b, s, 128] -----------------
#pragma unroll
  for (int r = 0; r < 8; ++r) {
    float inv = 1.0f / lrow[r];
    int s = q0 + r + hi * 8;
    size_t base = ((size_t)b * SEQ + s) * HDIM + h * DK;
    ctx16[base + lo]      = (half_t)(ctx0[r] * inv);
    ctx16[base + 16 + lo] = (half_t)(ctx1[r] * inv);
  }
}

// ---------------------------------------------------------------------------
// 3) output projection + residual + LayerNorm, fused.
//    grid = ROWS/16, block = 256 (8 waves).  Block owns 16 rows x 512 cols.
// ---------------------------------------------------------------------------
__global__ __launch_bounds__(256) void oproj_ln_kernel(
    const half_t* __restrict__ ctx16, const half_t* __restrict__ wto,
    const float* __restrict__ resid, const float* __restrict__ gamma,
    const float* __restrict__ beta, float* __restrict__ out) {
  __shared__ float rsum[16], rsq[16];

  const int m0 = blockIdx.x * 16;
  const int tid = threadIdx.x;
  const int wave = tid >> 5;
  const int lane = tid & 31;
  const int lo = lane & 15, hi = lane >> 4;

  if (tid < 16) { rsum[tid] = 0.f; rsq[tid] = 0.f; }
  __syncthreads();

  v8f acc[4];
#pragma unroll
  for (int t = 0; t < 4; ++t)
#pragma unroll
    for (int r = 0; r < 8; ++r) acc[t][r] = 0.f;

  const half_t* arow = ctx16 + (size_t)(m0 + lo) * HDIM + hi * 16;
#pragma unroll
  for (int kk = 0; kk < HDIM; kk += 32) {
    v16h a = pack16(*(const v8h*)(arow + kk), *(const v8h*)(arow + kk + 8));
#pragma unroll
    for (int t = 0; t < 4; ++t) {
      int n0 = (wave * 4 + t) * 16;
      const half_t* bp = wto + (size_t)(n0 + lo) * HDIM + kk + hi * 16;
      v16h bb = pack16(*(const v8h*)bp, *(const v8h*)(bp + 8));
      acc[t] = wmma_f16(a, bb, acc[t]);
    }
  }

  // residual add + per-row partial stats
  float x[4][8];
#pragma unroll
  for (int t = 0; t < 4; ++t) {
    int n = (wave * 4 + t) * 16 + lo;
#pragma unroll
    for (int r = 0; r < 8; ++r) {
      int m = m0 + r + hi * 8;
      x[t][r] = acc[t][r] + resid[(size_t)m * DMODEL + n];
    }
  }
#pragma unroll
  for (int r = 0; r < 8; ++r) {
    float s = 0.f, q = 0.f;
#pragma unroll
    for (int t = 0; t < 4; ++t) { s += x[t][r]; q += x[t][r] * x[t][r]; }
    atomicAdd(&rsum[r + hi * 8], s);
    atomicAdd(&rsq[r + hi * 8], q);
  }
  __syncthreads();

  const float invD = 1.0f / (float)DMODEL;
#pragma unroll
  for (int t = 0; t < 4; ++t) {
    int n = (wave * 4 + t) * 16 + lo;
    float g = gamma[n], bt = beta[n];
#pragma unroll
    for (int r = 0; r < 8; ++r) {
      int m = m0 + r + hi * 8;
      float mu = rsum[r + hi * 8] * invD;
      float var = rsq[r + hi * 8] * invD - mu * mu;
      out[(size_t)m * DMODEL + n] =
          (x[t][r] - mu) * rsqrtf(var + 1e-5f) * g + bt;
    }
  }
}

// ---------------------------------------------------------------------------
// host launch
// ---------------------------------------------------------------------------
extern "C" void kernel_launch(void* const* d_in, const int* in_sizes, int n_in,
                              void* d_out, int out_size, void* d_ws,
                              size_t ws_size, hipStream_t stream) {
  const float* in_q = (const float*)d_in[0];
  const float* in_k = (const float*)d_in[1];
  const float* in_v = (const float*)d_in[2];
  const float* w_q  = (const float*)d_in[3];
  const float* w_k  = (const float*)d_in[4];
  const float* w_v  = (const float*)d_in[5];
  const float* w_o  = (const float*)d_in[6];
  const float* ln_g = (const float*)d_in[7];
  const float* ln_b = (const float*)d_in[8];
  float* out = (float*)d_out;

  half_t* ws = (half_t*)d_ws;   // offsets in halves
  const size_t ACT = (size_t)ROWS * DMODEL;      // 8,388,608
  const size_t WSZ = (size_t)DMODEL * HDIM;      // 65,536
  const size_t HEADSZ = (size_t)BATCH * NHEAD * SEQ * DK;  // 2,097,152

  half_t* xq16  = ws;
  half_t* xk16  = xq16 + ACT;
  half_t* xv16  = xk16 + ACT;
  half_t* wtq16 = xv16 + ACT;
  half_t* wtk16 = wtq16 + WSZ;
  half_t* wtv16 = wtk16 + WSZ;
  half_t* wto16 = wtv16 + WSZ;
  half_t* q16   = wto16 + WSZ;
  half_t* k16   = q16 + HEADSZ;
  half_t* vT16  = k16 + HEADSZ;
  half_t* ctx16 = vT16 + HEADSZ;

  // 0) convert activations + weights
  {
    int n = (int)ACT;
    int blocks = n / (256 * 8);
    f32_to_f16_kernel<<<blocks, 256, 0, stream>>>(in_q, xq16, n);
    f32_to_f16_kernel<<<blocks, 256, 0, stream>>>(in_k, xk16, n);
    f32_to_f16_kernel<<<blocks, 256, 0, stream>>>(in_v, xv16, n);
  }
  {
    int n = (int)WSZ;
    int blocks = (n + 255) / 256;
    w_transpose_kernel<<<blocks, 256, 0, stream>>>(w_q, wtq16, DMODEL, HDIM);
    w_transpose_kernel<<<blocks, 256, 0, stream>>>(w_k, wtk16, DMODEL, HDIM);
    w_transpose_kernel<<<blocks, 256, 0, stream>>>(w_v, wtv16, DMODEL, HDIM);
    w_transpose_kernel<<<blocks, 256, 0, stream>>>(w_o, wto16, HDIM, DMODEL);
  }

  // 1) QKV projection (16x64 strip per wave)
  {
    dim3 grid(ROWS / 16, HDIM / 64, 3);
    qkv_proj_kernel<<<grid, 32, 0, stream>>>(xq16, xk16, xv16, wtq16, wtk16,
                                             wtv16, q16, k16, vT16);
  }

  // 2) flash attention (4 waves / block, LDS-staged K/V, double buffered)
  {
    dim3 grid(SEQ / 64, BATCH * NHEAD);
    flash_attn_kernel<<<grid, 128, 0, stream>>>(q16, k16, vT16, ctx16);
  }

  // 3) output projection + residual + LayerNorm
  {
    oproj_ln_kernel<<<ROWS / 16, 256, 0, stream>>>(ctx16, wto16, in_q, ln_g,
                                                   ln_b, out);
  }
}
